// GINEncoder_80642305950442
// MI455X (gfx1250) — matrix operand order repaired and checked by reading, not compiled, not run on hardware
//
#include <hip/hip_runtime.h>
#include <hip/hip_bf16.h>

#define NN 50000
#define NE 800000
#define NG 256
#define D  256
#define NL 3
#define BN_EPS 1e-5f
#define RPB 64

typedef __attribute__((ext_vector_type(16))) __bf16 v16bf;
typedef __attribute__((ext_vector_type(8)))  float  v8f;

__global__ void zero_f32(float* p, int n) {
  int i = blockIdx.x * blockDim.x + threadIdx.x;
  if (i < n) p[i] = 0.0f;
}

__global__ void copy_f32(const float4* __restrict__ src, float4* __restrict__ dst, int n4) {
  int i = blockIdx.x * blockDim.x + threadIdx.x;
  int stride = gridDim.x * blockDim.x;
  for (; i < n4; i += stride) dst[i] = src[i];
}

// agg[dst[e]][c..c+3] += h[src[e]][c..c+3]; one thread per (edge, 4-feature chunk)
__global__ void edge_scatter(const float* __restrict__ h, const int* __restrict__ src,
                             const int* __restrict__ dst, float* __restrict__ agg) {
  int gid = blockIdx.x * blockDim.x + threadIdx.x;
  int e = gid >> 6;
  int c = (gid & 63) << 2;
  if (e >= NE) return;
  int s = src[e], d = dst[e];
  const float4 v = *(const float4*)(h + (size_t)s * D + c);
  float* a = agg + (size_t)d * D + c;
  atomicAdd(a + 0, v.x);
  atomicAdd(a + 1, v.y);
  atomicAdd(a + 2, v.z);
  atomicAdd(a + 3, v.w);
}

// Pack w1/w2 (f32 row-major [K=256][N=256]) into bf16 WMMA B-fragment order:
// block per (layer,mat,tn,tk): 32 lanes x 16 elems; lane half selects K+8 within each 16-K half,
// element idx -> K = tk*32 + (idx<8 ? half*8 + idx : 16 + half*8 + idx-8), N = tn*16 + (lane&15).
__global__ void prepack_w(const float* __restrict__ w1, const float* __restrict__ w2,
                          __bf16* __restrict__ wpk) {
  int tid = blockIdx.x * blockDim.x + threadIdx.x;
  if (tid >= NL * 2 * 16 * 8 * 32) return;
  int lane = tid & 31;
  int tk   = (tid >> 5) & 7;
  int tn   = (tid >> 8) & 15;
  int m    = (tid >> 12) & 1;
  int i    = tid >> 13;
  const float* W = (m ? w2 : w1) + (size_t)i * D * D;
  int half = lane >> 4;
  int n = tn * 16 + (lane & 15);
  __bf16* out = wpk + (size_t)tid * 16;
#pragma unroll
  for (int idx = 0; idx < 16; ++idx) {
    int k = tk * 32 + (idx < 8 ? (half << 3) + idx : 16 + (half << 3) + (idx - 8));   // B 32x16 layout: lanes16-31 -> K+8 within each 16-K half
    out[idx] = (__bf16)W[k * D + n];
  }
}

// Out[M,N] = relu( A[M,K] * Wpk + bias ), M=NN, N=K=256.
// 8 waves: waveM in {0,1} x 16 rows, waveN in {0..3} x 64 cols. 32 WMMAs/wave.
__global__ __launch_bounds__(256)
void gemm_bias_relu(const float* __restrict__ A, const __bf16* __restrict__ wp,
                    const float* __restrict__ bias, float* __restrict__ Out) {
  const int lane  = threadIdx.x & 31;
  const int wave  = threadIdx.x >> 5;
  const int waveM = wave >> 2;
  const int waveN = wave & 3;
  const int rowBase = blockIdx.x * 32 + waveM * 16;
  const int nBase   = waveN * 64;
  const int half    = lane >> 4;
  const int r       = rowBase + (lane & 15);

  v8f acc[4];
#pragma unroll
  for (int t = 0; t < 4; ++t)
#pragma unroll
    for (int v = 0; v < 8; ++v) acc[t][v] = 0.0f;

#pragma unroll
  for (int kt = 0; kt < 8; ++kt) {
    // A fragment, 16-bit 16x32 layout: lane l (M=l&15), VGPR j elem e:
    // K = (j>>2)*16 + half*8 + (j&3)*2 + e  -> two contiguous 8-float runs.
    v16bf af;
#pragma unroll
    for (int q = 0; q < 16; ++q) af[q] = (__bf16)0.0f;
    if (r < NN) {
      const float* ap = A + (size_t)r * D + kt * 32 + half * 8;
      float4 a0 = *(const float4*)(ap);
      float4 a1 = *(const float4*)(ap + 4);
      float4 a2 = *(const float4*)(ap + 16);
      float4 a3 = *(const float4*)(ap + 20);
      af[0]  = (__bf16)a0.x; af[1]  = (__bf16)a0.y; af[2]  = (__bf16)a0.z; af[3]  = (__bf16)a0.w;
      af[4]  = (__bf16)a1.x; af[5]  = (__bf16)a1.y; af[6]  = (__bf16)a1.z; af[7]  = (__bf16)a1.w;
      af[8]  = (__bf16)a2.x; af[9]  = (__bf16)a2.y; af[10] = (__bf16)a2.z; af[11] = (__bf16)a2.w;
      af[12] = (__bf16)a3.x; af[13] = (__bf16)a3.y; af[14] = (__bf16)a3.z; af[15] = (__bf16)a3.w;
    }
#pragma unroll
    for (int t = 0; t < 4; ++t) {
      int tn = (nBase >> 4) + t;
      const v16bf bf = *(const v16bf*)(wp + (size_t)((tn * 8 + kt) * 32 + lane) * 16);
      acc[t] = __builtin_amdgcn_wmma_f32_16x16x32_bf16(
          false, af, false, bf, (short)0, acc[t], false, false);
    }
  }

  // C/D layout: VGPR v -> M = rowBase + v + half*8, N = nBase + t*16 + (lane&15)
#pragma unroll
  for (int t = 0; t < 4; ++t) {
    int Ncol = nBase + t * 16 + (lane & 15);
    float bv = bias[Ncol];
#pragma unroll
    for (int v = 0; v < 8; ++v) {
      int M = rowBase + v + half * 8;
      if (M < NN) {
        float val = acc[t][v] + bv;
        Out[(size_t)M * D + Ncol] = fmaxf(val, 0.0f);
      }
    }
  }
}

// Per-feature sum & sumsq over all nodes -> stats[0..255]=sum, stats[256..511]=sumsq
__global__ void bn_stats(const float* __restrict__ z, float* __restrict__ stats) {
  int f = threadIdx.x;
  float s = 0.0f, q = 0.0f;
  for (int r = blockIdx.x; r < NN; r += gridDim.x) {
    float v = z[(size_t)r * D + f];
    s += v;
    q += v * v;
  }
  atomicAdd(&stats[f], s);
  atomicAdd(&stats[D + f], q);
}

// BN apply in place (z becomes next-layer h) + pooled segment-sum into out.
// batch is sorted -> run-length accumulate per graph, one atomic per run.
__global__ void bn_apply_pool(float* __restrict__ z, const int* __restrict__ batch,
                              const float* __restrict__ gamma, const float* __restrict__ beta,
                              const float* __restrict__ stats, float* __restrict__ out,
                              int layerOff) {
  int f = threadIdx.x;
  int n0 = blockIdx.x * RPB;
  if (n0 >= NN) return;
  float inv   = 1.0f / (float)NN;
  float mean  = stats[f] * inv;
  float var   = stats[D + f] * inv - mean * mean;
  float scale = gamma[f] * rsqrtf(var + BN_EPS);
  float shift = beta[f] - mean * scale;
  int nEnd = n0 + RPB;
  if (nEnd > NN) nEnd = NN;
  float acc = 0.0f;
  int g = batch[n0];
  for (int n = n0; n < nEnd; ++n) {
    int gn = batch[n];
    if (gn != g) {
      atomicAdd(&out[(size_t)g * (NL * D) + layerOff + f], acc);
      acc = 0.0f;
      g = gn;
    }
    float v = z[(size_t)n * D + f] * scale + shift;
    z[(size_t)n * D + f] = v;
    acc += v;
  }
  atomicAdd(&out[(size_t)g * (NL * D) + layerOff + f], acc);
}

extern "C" void kernel_launch(void* const* d_in, const int* in_sizes, int n_in,
                              void* d_out, int out_size, void* d_ws, size_t ws_size,
                              hipStream_t stream) {
  const float* x     = (const float*)d_in[0];
  const int*   ei    = (const int*)d_in[1];
  const int*   batch = (const int*)d_in[2];
  const float* w1    = (const float*)d_in[3];
  const float* b1    = (const float*)d_in[4];
  const float* w2    = (const float*)d_in[5];
  const float* b2    = (const float*)d_in[6];
  const float* gamma = (const float*)d_in[7];
  const float* beta  = (const float*)d_in[8];
  float* out = (float*)d_out;

  float*  bufA  = (float*)d_ws;
  float*  bufB  = bufA + (size_t)NN * D;
  float*  stats = bufB + (size_t)NN * D;
  __bf16* wpk   = (__bf16*)(stats + 2 * D);

  const int* src = ei;
  const int* dst = ei + NE;

  zero_f32<<<(NG * NL * D + 255) / 256, 256, 0, stream>>>(out, NG * NL * D);
  prepack_w<<<(NL * 2 * 16 * 8 * 32 + 255) / 256, 256, 0, stream>>>(w1, w2, wpk);

  for (int i = 0; i < NL; ++i) {
    const float* hin = (i == 0) ? x : ((i == 1) ? bufA : bufB);
    float* P = (i == 1) ? bufB : bufA;  // agg, then GEMM2 output (BN in place)
    float* Q = (i == 1) ? bufA : bufB;  // GEMM1 output (hidden)

    // P = hin  (GIN eps=0 self term), then scatter-add neighbors
    copy_f32<<<4096, 256, 0, stream>>>((const float4*)hin, (float4*)P, NN * D / 4);
    edge_scatter<<<(NE * 64) / 256, 256, 0, stream>>>(hin, src, dst, P);

    zero_f32<<<2, 256, 0, stream>>>(stats, 2 * D);

    gemm_bias_relu<<<(NN + 31) / 32, 256, 0, stream>>>(
        P, wpk + (size_t)(i * 2 + 0) * 65536, b1 + i * D, Q);
    gemm_bias_relu<<<(NN + 31) / 32, 256, 0, stream>>>(
        Q, wpk + (size_t)(i * 2 + 1) * 65536, b2 + i * D, P);

    bn_stats<<<512, 256, 0, stream>>>(P, stats);
    bn_apply_pool<<<(NN + RPB - 1) / RPB, 256, 0, stream>>>(
        P, batch, gamma + i * D, beta + i * D, stats, out, i * D);
  }
}
